// MixedLoss_64957085385469
// MI455X (gfx1250) — compile-verified, matrix operand order checked by reading, not run on hardware
//
#include <hip/hip_runtime.h>

typedef __attribute__((ext_vector_type(16))) __bf16 v16bf;
typedef __attribute__((ext_vector_type(8)))  __bf16 v8bf;
typedef __attribute__((ext_vector_type(8)))  float  v8f;

#define NROWS 4096
#define DIM   256
#define NCLS  128
#define KSHOT 16
#define QSHOT 16
#define SUP   2048
#define INVT  10.0f      // 1/temperature
#define ALPHA 0.5f

__device__ __forceinline__ unsigned short f2bf(float f) {
  unsigned u = __float_as_uint(f);
  unsigned r = (u + 0x7FFFu + ((u >> 16) & 1u)) >> 16;
  return (unsigned short)r;
}

// ISA 7.12.2 16-bit A/B fragment: lane holds row (lane&15); K-halves split by lane>>4.
// Elements 0..7 -> K = kb..kb+7 ; elements 8..15 -> K = kb+16..kb+23 ; kb = (lane>>4)*8 + kk*32.
__device__ __forceinline__ v16bf load_frag(const unsigned short* rowPtr, int kk, int lane) {
  const __bf16* p = (const __bf16*)rowPtr + kk * 32 + ((lane >> 4) << 3);
  v8bf lo = *(const v8bf*)p;
  v8bf hi = *(const v8bf*)(p + 16);
  return __builtin_shufflevector(lo, hi, 0,1,2,3,4,5,6,7,8,9,10,11,12,13,14,15);
}

// ---------------- prep: integrate over repeat, norms, bf16 copies ----------------
__global__ void prep_features(const float* __restrict__ tasks, float* __restrict__ F,
                              float* __restrict__ x2, unsigned short* __restrict__ Fbf,
                              unsigned short* __restrict__ Fnbf) {
  __shared__ float sred[256];
  int i = blockIdx.x, k = threadIdx.x;
  float f = 0.5f * (tasks[(size_t)i * DIM + k] +
                    tasks[(size_t)NROWS * DIM + (size_t)i * DIM + k]);
  F[(size_t)i * DIM + k] = f;
  sred[k] = f * f;
  __syncthreads();
  for (int s = 128; s > 0; s >>= 1) { if (k < s) sred[k] += sred[k + s]; __syncthreads(); }
  float ss = sred[0];
  if (k == 0) x2[i] = ss;
  float rin = rsqrtf(fmaxf(ss, 1e-30f));
  Fbf[(size_t)i * DIM + k]  = f2bf(f);
  Fnbf[(size_t)i * DIM + k] = f2bf(f * rin);
}

__global__ void prep_protos(const float* __restrict__ F, float* __restrict__ P,
                            float* __restrict__ p2, unsigned short* __restrict__ Pbf) {
  __shared__ float sred[256];
  int c = blockIdx.x, k = threadIdx.x;
  float s = 0.f;
  #pragma unroll
  for (int j = 0; j < KSHOT; ++j) s += F[(size_t)(c * KSHOT + j) * DIM + k];
  s *= (1.0f / KSHOT);
  P[(size_t)c * DIM + k] = s;
  Pbf[(size_t)c * DIM + k] = f2bf(s);
  sred[k] = s * s;
  __syncthreads();
  for (int t = 128; t > 0; t >>= 1) { if (k < t) sred[k] += sred[k + t]; __syncthreads(); }
  if (k == 0) p2[c] = sred[0];
}

// ---------------- dists (WMMA) + softmax(-D) -> prob, proto loss, acc ----------------
__global__ void dists_prob(const unsigned short* __restrict__ Fbf,
                           const unsigned short* __restrict__ Pbf,
                           const float* __restrict__ x2, const float* __restrict__ p2,
                           float* __restrict__ prob, float* __restrict__ scal) {
  __shared__ float lds_d[16 * 128];
  __shared__ float lds_x2[16];
  __shared__ float lds_p2[128];
  int tid = threadIdx.x;
  int wave = tid >> 5, lane = tid & 31, hi = lane >> 4;
  int rowBase = blockIdx.x * 16;
  if (tid < 16)  lds_x2[tid] = x2[rowBase + tid];
  if (tid < 128) lds_p2[tid] = p2[tid];
  const unsigned short* aRow = Fbf + (size_t)(rowBase + (lane & 15)) * DIM;
  v16bf afrag[8];
  #pragma unroll
  for (int kk = 0; kk < 8; ++kk) afrag[kk] = load_frag(aRow, kk, lane);
  __syncthreads();
  #pragma unroll
  for (int t = 0; t < 2; ++t) {
    int ct = wave * 2 + t;
    const unsigned short* bRow = Pbf + (size_t)(ct * 16 + (lane & 15)) * DIM;
    v8f c = {};
    #pragma unroll
    for (int kk = 0; kk < 8; ++kk) {
      v16bf b = load_frag(bRow, kk, lane);
      c = __builtin_amdgcn_wmma_f32_16x16x32_bf16(false, afrag[kk], false, b,
                                                  (short)0, c, false, false);
    }
    int n = ct * 16 + (lane & 15);
    #pragma unroll
    for (int r = 0; r < 8; ++r) {
      int m = r + 8 * hi;
      lds_d[m * 128 + n] = lds_x2[m] + lds_p2[n] - 2.0f * c[r];
    }
  }
  __syncthreads();
  int row = tid >> 3, sub = tid & 7;
  const float* dr = &lds_d[row * 128];
  float minv = 3.4e38f; int mini = 0;
  #pragma unroll
  for (int j = 0; j < 16; ++j) {
    int cidx = sub * 16 + j;
    float v = dr[cidx];
    if (v < minv) { minv = v; mini = cidx; }
  }
  for (int off = 1; off < 8; off <<= 1) {
    float ov = __shfl_xor(minv, off);
    int   oi = __shfl_xor(mini, off);
    if (ov < minv || (ov == minv && oi < mini)) { minv = ov; mini = oi; }
  }
  float s = 0.f;
  #pragma unroll
  for (int j = 0; j < 16; ++j) s += __expf(minv - dr[sub * 16 + j]);
  for (int off = 1; off < 8; off <<= 1) s += __shfl_xor(s, off);
  float inv = 1.0f / s;
  int gi = rowBase + row;
  #pragma unroll
  for (int j = 0; j < 16; ++j) {
    int cidx = sub * 16 + j;
    prob[(size_t)gi * NCLS + cidx] = __expf(minv - dr[cidx]) * inv;
  }
  if (gi >= SUP) {
    int lab = (gi - SUP) >> 4;
    if (sub == (lab >> 4))
      unsafeAtomicAdd(&scal[0], (minv - dr[lab]) - __logf(s));
    if (sub == 0)
      unsafeAtomicAdd(&scal[1], (mini == lab) ? 1.0f : 0.0f);
  }
}

// ---------------- pairwise prototype loss (single block) ----------------
__global__ void pairwise(const float* __restrict__ P, float* __restrict__ sq,
                         float* __restrict__ scal) {
  __shared__ float sred[256];
  int tid = threadIdx.x;
  for (int idx = tid; idx < NCLS * NCLS; idx += 256) {
    int a = idx >> 7, b = idx & 127;
    const float* pa = &P[(size_t)a * DIM];
    const float* pb = &P[(size_t)b * DIM];
    float s = 0.f;
    for (int k = 0; k < DIM; ++k) { float d = pa[k] - pb[k]; s += d * d; }
    sq[idx] = s * (1.0f / 16.0f);
  }
  __syncthreads();
  float psum = 0.f, pcnt = 0.f;
  for (int idx = tid; idx < NCLS * NCLS; idx += 256) {
    float v = sq[idx];
    if (v != 0.f) { psum += v; pcnt += 1.f; }
  }
  sred[tid] = psum; __syncthreads();
  for (int t = 128; t > 0; t >>= 1) { if (tid < t) sred[tid] += sred[tid + t]; __syncthreads(); }
  float total = sred[0]; __syncthreads();
  sred[tid] = pcnt; __syncthreads();
  for (int t = 128; t > 0; t >>= 1) { if (tid < t) sred[tid] += sred[tid + t]; __syncthreads(); }
  float n = sred[0]; __syncthreads();
  float mean = total / n;
  float pv = 0.f;
  for (int idx = tid; idx < NCLS * NCLS; idx += 256) {
    float v = sq[idx];
    if (v != 0.f) { float d = v - mean; pv += d * d; }
  }
  sred[tid] = pv; __syncthreads();
  for (int t = 128; t > 0; t >>= 1) { if (tid < t) sred[tid] += sred[tid + t]; __syncthreads(); }
  float var = sred[0] / (n - 1.0f); __syncthreads();
  float istd = rsqrtf(var);
  float w = 0.f;
  for (int idx = tid; idx < NCLS * NCLS; idx += 256) {
    int a = idx >> 7, b = idx & 127;
    if (a != b) w += __expf(-sq[idx] * istd);
  }
  sred[tid] = w; __syncthreads();
  for (int t = 128; t > 0; t >>= 1) { if (tid < t) sred[tid] += sred[tid + t]; __syncthreads(); }
  if (tid == 0) scal[3] = sred[0] / (float)(NCLS * NCLS - NCLS);
}

// ---------------- contrastive loss: single pass, online softmax, async B staging ----------------
// Block = 32 rows (2 row-tiles) x 4 waves:
//   wave0: rows 0-15, ct 0..127     wave1: rows 16-31, ct 0..127
//   wave2: rows 0-15, ct 128..255   wave3: rows 16-31, ct 128..255
// Two 8KB B tiles staged per iteration via global_load_async_to_lds_b128 (double buffered).
__global__ void __launch_bounds__(128)
contrastive(const unsigned short* __restrict__ Fnbf,
            const float* __restrict__ prob, float* __restrict__ scal) {
  __shared__ unsigned short bbuf[2][2][16 * 256];  // [buffer][ctHalf][16x256 bf16] = 32KB
  __shared__ float lds_prob[32 * 128];             // 16KB
  __shared__ float red[4][16][6];
  int tid = threadIdx.x, wave = tid >> 5, lane = tid & 31, hi = lane >> 4;
  int rowBase = blockIdx.x * 32;
  int rowTile = wave & 1;
  int ctHalf  = wave >> 1;
  int wrow = rowBase + rowTile * 16;
  int labI = (wrow & 2047) >> 4;                   // label uniform per 16-row tile

  for (int idx = tid; idx < 32 * 128; idx += 128)
    lds_prob[idx] = prob[(size_t)rowBase * NCLS + idx];

  const unsigned short* aRow = Fnbf + (size_t)(wrow + (lane & 15)) * DIM;
  v16bf afrag[8];
  #pragma unroll
  for (int kk = 0; kk < 8; ++kk) afrag[kk] = load_frag(aRow, kk, lane);

  // LDS byte offsets of the two B buffers (flat addr low 32 bits = LDS offset)
  unsigned bb0 = (unsigned)(uintptr_t)&bbuf[0][0][0];
  unsigned bb1 = (unsigned)(uintptr_t)&bbuf[1][0][0];
  // staging: thread covers (tid>>6)-th ctHalf tile; 8 x b128 chunks each (64 thr x 8 x 16B = 8KB)
  int sHalf = tid >> 6, sLt = tid & 63;

  // online-softmax state per r (8 rows of my tile)
  float m[8], S1[8], See[8], Snep[8], Sme[8], smz[8];
  #pragma unroll
  for (int r = 0; r < 8; ++r) { m[r] = -3.4e38f; S1[r] = 0.f; See[r] = 0.f;
                                Snep[r] = 0.f;   Sme[r] = 0.f; smz[r] = 0.f; }

  // prologue: stage iteration 0 into buffer 0
  {
    uint64_t sa = (uint64_t)(uintptr_t)(Fnbf + (size_t)(sHalf * 128) * 16 * DIM);
    #pragma unroll
    for (int j = 0; j < 8; ++j) {
      unsigned off = (unsigned)(sLt * 16 + j * 1024);
      unsigned lds = bb0 + (unsigned)sHalf * 8192u + off;
      asm volatile("global_load_async_to_lds_b128 %0, %1, %2"
                   :: "v"(lds), "v"(off), "s"(sa) : "memory");
    }
  }

  for (int it = 0; it < 128; ++it) {
    int cur = it & 1;
    if (it + 1 < 128) {
      unsigned bbn = cur ? bb0 : bb1;
      uint64_t sa = (uint64_t)(uintptr_t)(Fnbf + (size_t)(sHalf * 128 + it + 1) * 16 * DIM);
      #pragma unroll
      for (int j = 0; j < 8; ++j) {
        unsigned off = (unsigned)(sLt * 16 + j * 1024);
        unsigned lds = bbn + (unsigned)sHalf * 8192u + off;
        asm volatile("global_load_async_to_lds_b128 %0, %1, %2"
                     :: "v"(lds), "v"(off), "s"(sa) : "memory");
      }
      asm volatile("s_wait_asynccnt 0x8" ::: "memory");  // in-order: current tile complete
    } else {
      asm volatile("s_wait_asynccnt 0x0" ::: "memory");
    }
    __syncthreads();  // all waves' staged chunks visible

    const unsigned short* bRow = &bbuf[cur][ctHalf][(lane & 15) * DIM];
    v8f c = {};
    #pragma unroll
    for (int kk = 0; kk < 8; ++kk) {
      v16bf b = load_frag(bRow, kk, lane);   // ds_load_b128 x2
      c = __builtin_amdgcn_wmma_f32_16x16x32_bf16(false, afrag[kk], false, b,
                                                  (short)0, c, false, false);
    }
    int ct = ctHalf * 128 + it;
    int labJ = ((ct * 16) & 2047) >> 4;
    int gj = ct * 16 + (lane & 15);
    bool same = (labJ == labI);
    #pragma unroll
    for (int r = 0; r < 8; ++r) {
      float z = c[r] * INVT;
      float e;
      if (z > m[r]) {                          // rare after warmup
        float f = __expf(m[r] - z);
        S1[r] *= f; See[r] *= f * f; Snep[r] *= f; Sme[r] *= f;
        m[r] = z; e = 1.0f;
      } else {
        e = __expf(z - m[r]);
      }
      if (same) {
        int giR = wrow + r + 8 * hi;
        if (giR != gj) { Sme[r] += e; smz[r] += z; }
      } else {
        S1[r]  += e;
        See[r] += e * e;
        Snep[r] += e * lds_prob[(rowTile * 16 + r + 8 * hi) * 128 + labJ];
      }
    }
    __syncthreads();  // done reading bbuf[cur] before it is overwritten
  }

  // intra-wave merge across the 16 lanes of each half (online-softmax merge)
  #pragma unroll
  for (int r = 0; r < 8; ++r) {
    for (int off = 1; off < 16; off <<= 1) {
      float om  = __shfl_xor(m[r],   off);
      float oS1 = __shfl_xor(S1[r],  off);
      float oSe = __shfl_xor(See[r], off);
      float oSn = __shfl_xor(Snep[r],off);
      float oSm = __shfl_xor(Sme[r], off);
      float mN  = fmaxf(m[r], om);
      float fs  = __expf(m[r] - mN), fo = __expf(om - mN);
      S1[r]   = S1[r] * fs        + oS1 * fo;
      See[r]  = See[r] * fs * fs  + oSe * fo * fo;
      Snep[r] = Snep[r] * fs      + oSn * fo;
      Sme[r]  = Sme[r] * fs       + oSm * fo;
      m[r] = mN;
      smz[r] += __shfl_xor(smz[r], off);
    }
    if ((lane & 15) == 0) {
      int rr = r + 8 * hi;
      red[wave][rr][0] = m[r];   red[wave][rr][1] = S1[r];
      red[wave][rr][2] = See[r]; red[wave][rr][3] = Snep[r];
      red[wave][rr][4] = Sme[r]; red[wave][rr][5] = smz[r];
    }
  }
  __syncthreads();
  // merge ct-halves: waves {t, t+2} hold the same row tile t
  if (tid < 32) {
    int t = tid >> 4, rr = tid & 15;
    int wA = t, wB = t + 2;
    float mA = red[wA][rr][0], mB = red[wB][rr][0];
    float mN = fmaxf(mA, mB);
    float fA = __expf(mA - mN), fB = __expf(mB - mN);
    float S1t = red[wA][rr][1] * fA      + red[wB][rr][1] * fB;
    float Set = red[wA][rr][2] * fA * fA + red[wB][rr][2] * fB * fB;
    float Snt = red[wA][rr][3] * fA      + red[wB][rr][3] * fB;
    float Smt = red[wA][rr][4] * fA      + red[wB][rr][4] * fB;
    float szt = red[wA][rr][5]           + red[wB][rr][5];
    float Z = ALPHA * Set / S1t + (1.0f - ALPHA) * Snt + Smt;
    float cnt = (float)(KSHOT + QSHOT - 1);   // 31 positives per row
    float mlpp = (szt - cnt * mN) / cnt - __logf(Z);
    unsafeAtomicAdd(&scal[2], mlpp);
  }
}

__global__ void init_scal(float* scal) { if (threadIdx.x < 16) scal[threadIdx.x] = 0.f; }

__global__ void finalize(const float* __restrict__ scal, float* __restrict__ out) {
  float loss_pn = -scal[0] / (float)(NCLS * QSHOT);
  float acc     =  scal[1] / (float)(NCLS * QSHOT);
  float con     = -scal[2] / (float)NROWS;
  out[0] = loss_pn + 0.1f * scal[3] + 0.1f * con;
  out[1] = acc;
}

extern "C" void kernel_launch(void* const* d_in, const int* in_sizes, int n_in,
                              void* d_out, int out_size, void* d_ws, size_t ws_size,
                              hipStream_t stream) {
  const float* tasks = (const float*)d_in[0];
  char* ws = (char*)d_ws;
  float* F    = (float*)(ws + 0);              // 4096*256 f32
  float* prob = (float*)(ws + 4194304);        // 4096*128 f32
  float* P    = (float*)(ws + 6291456);        // 128*256 f32
  float* x2   = (float*)(ws + 6422528);        // 4096 f32
  float* p2   = (float*)(ws + 6438912);        // 128 f32
  float* scal = (float*)(ws + 6439424);        // 16 f32 accumulators
  float* sq   = (float*)(ws + 6439488);        // 128*128 f32
  unsigned short* Fbf  = (unsigned short*)(ws + 6505024);   // 4096*256 bf16
  unsigned short* Fnbf = (unsigned short*)(ws + 8602176);   // 4096*256 bf16 (normalized)
  unsigned short* Pbf  = (unsigned short*)(ws + 10699328);  // 128*256 bf16

  init_scal<<<1, 32, 0, stream>>>(scal);
  prep_features<<<NROWS, 256, 0, stream>>>(tasks, F, x2, Fbf, Fnbf);
  prep_protos<<<NCLS, 256, 0, stream>>>(F, P, p2, Pbf);
  dists_prob<<<NROWS / 16, 128, 0, stream>>>(Fbf, Pbf, x2, p2, prob, scal);
  pairwise<<<1, 256, 0, stream>>>(P, sq, scal);
  contrastive<<<NROWS / 32, 128, 0, stream>>>(Fnbf, prob, scal);
  finalize<<<1, 1, 0, stream>>>(scal, (float*)d_out);
}
